// FCA_Block_25254407701037
// MI455X (gfx1250) — matrix operand order repaired and checked by reading, not compiled, not run on hardware
//
#include <hip/hip_runtime.h>
#include <hip/hip_bf16.h>

// ---------------------------------------------------------------------------
// Fused FCA block for MI455X (gfx1250), wave32 + v_wmma_f32_16x16x32_bf16,
// with TDM (tensor_load_to_lds) weight staging.
//
// prep kernel: w1,w2 f32 -> bf16 TRANSPOSED ([n][k]) in workspace, plus the
//              4*cos DCT matrix as bf16.  Amortized across all 2048 WGs.
// main kernel: one WG (8 wave32) per (b,h) slice:
//   dct = Dm[64x64] @ x            (K=64,  2 wmma k-steps)
//   h   = relu(dct @ w1 + b1)      (K=256, 8 wmma k-steps)
//   wgt = sigmoid(h @ w2 + b2)     (K=256, 8 wmma k-steps)
//   out = x * wgt
// Weights land in LDS column-major (padded stride) via ONE tensor_load_to_lds
// per matrix, using the D# pad fields to insert the 16B/row padding.
// ---------------------------------------------------------------------------

typedef __attribute__((ext_vector_type(8)))  __bf16 v8bf;
typedef __attribute__((ext_vector_type(16))) __bf16 v16bf;
typedef __attribute__((ext_vector_type(8)))  float  v8f;
typedef __attribute__((ext_vector_type(4)))  float  f32x4;
typedef __attribute__((ext_vector_type(4)))  unsigned int u32x4;
typedef __attribute__((ext_vector_type(8)))  int    i32x8;
typedef __attribute__((ext_vector_type(4)))  int    i32x4;

// LDS strides in bf16 elements (padded: 16B-aligned rows, spread banks)
#define SA_STR 72     // Dm: 64 x 64  row-major
#define SX_STR 72     // x : 256 cols x 64 rows, column-major (B operand)
#define SD_STR 264    // dct/h: 64 x 256 row-major (A operand)
#define SW_STR 264    // w : 256 cols x 256 rows, column-major (B operand)

// byte offsets inside dynamic LDS
#define OFF_A 0                         // 64*72*2   = 9216
#define OFF_X 9216                      // 256*72*2  = 36864
#define OFF_D 46080                     // 64*264*2  = 33792
#define OFF_H 79872                     // 64*264*2  = 33792
#define OFF_W 113664                    // 256*264*2 = 135168
#define LDS_BYTES 248832

// workspace layout (bytes)
#define WS_W1T 0         // 65536 bf16 = 131072
#define WS_W2T 131072    // 65536 bf16 = 131072
#define WS_DM  262144    //  4096 bf16 =   8192

#if __has_builtin(__builtin_amdgcn_tensor_load_to_lds) && \
    __has_builtin(__builtin_amdgcn_s_wait_tensorcnt)
#define HAVE_TDM 1
#else
#define HAVE_TDM 0
#endif

__device__ __forceinline__ v8f vzero8() {
  v8f z;
#pragma unroll
  for (int i = 0; i < 8; ++i) z[i] = 0.0f;
  return z;
}

// A fragment: 16x32 bf16 tile from row-major LDS matrix.
__device__ __forceinline__ v16bf load_a(const __bf16* A, int ld, int mBase,
                                        int kBase, int lane) {
  const int half = lane >> 4, m = lane & 15;
  const __bf16* row = A + (size_t)(mBase + m) * ld + kBase + half * 8;
  v8bf lo = *(const v8bf*)(row);
  v8bf hi = *(const v8bf*)(row + 16);
  return __builtin_shufflevector(lo, hi, 0, 1, 2, 3, 4, 5, 6, 7,
                                 8, 9, 10, 11, 12, 13, 14, 15);
}

// B fragment: 32x16 bf16 tile from COLUMN-major LDS matrix (ld = column stride).
__device__ __forceinline__ v16bf load_b(const __bf16* Bcm, int ld, int kBase,
                                        int nBase, int lane) {
  const int half = lane >> 4, n = nBase + (lane & 15);
  const __bf16* col = Bcm + (size_t)n * ld + kBase + half * 8;
  v8bf lo = *(const v8bf*)(col);
  v8bf hi = *(const v8bf*)(col + 16);
  return __builtin_shufflevector(lo, hi, 0, 1, 2, 3, 4, 5, 6, 7,
                                 8, 9, 10, 11, 12, 13, 14, 15);
}

__device__ __forceinline__ v8f wmma_bf16(v16bf a, v16bf b, v8f c) {
  return __builtin_amdgcn_wmma_f32_16x16x32_bf16(
      false, a, false, b, (short)0, c, false, false);
}

#if HAVE_TDM
// One 2D TDM: copy tile_n rows of tile_k bf16 (row stride = stride_k elements
// in global) into LDS at lds_off, inserting (pad_amount) DWORDs of padding
// after every (pad_interval) DWORDs -> builds the padded column-major image.
__device__ __forceinline__ void tdm_load_2d(const void* gptr, unsigned lds_off,
                                            unsigned tile_k, unsigned tile_n,
                                            unsigned stride_k,
                                            unsigned pad_interval_code,
                                            unsigned pad_amount_code) {
  unsigned long long ga = (unsigned long long)gptr;
  u32x4 g0;
  g0[0] = 1u;                                              // count=1, user D#
  g0[1] = lds_off;                                         // lds_addr
  g0[2] = (unsigned)(ga & 0xFFFFFFFFu);                    // global_addr[31:0]
  g0[3] = (unsigned)((ga >> 32) & 0x1FFFFFFu) | (2u << 30);// addr[56:32]|type=2
  i32x8 g1;
  g1[0] = (int)((1u << 16) |                 // data_size = 2 bytes
                (1u << 20) |                 // pad_enable
                (pad_interval_code << 22) | (pad_amount_code << 25));
  g1[1] = (int)((tile_k & 0xFFFFu) << 16);   // tensor_dim0[15:0]
  g1[2] = (int)((tile_n & 0xFFFFu) << 16);   // dim0[31:16]=0, tensor_dim1[15:0]
  g1[3] = (int)((tile_k & 0xFFFFu) << 16);   // dim1[31:16]=0, tile_dim0
  g1[4] = (int)(tile_n & 0xFFFFu);           // tile_dim1, tile_dim2=0
  g1[5] = (int)stride_k;                     // tensor_dim0_stride[31:0]
  g1[6] = 0;                                 // stride0[47:32], dim1_stride lo
  g1[7] = 0;
  i32x4 g2 = {0, 0, 0, 0};
  i32x4 g3 = {0, 0, 0, 0};
  i32x8 g4 = {0, 0, 0, 0, 0, 0, 0, 0};       // unused extra group (6-arg form)
  __builtin_amdgcn_tensor_load_to_lds(g0, g1, g2, g3, g4, 0);
}
#endif

// stage weight matrix (pre-transposed bf16 [n][k], k contiguous) into LDS
// column-major with SW_STR padding.  Row = 256 bf16 = 512B = 128 DW -> pad
// code 6; pad = 8 bf16 = 16B = 4 DW -> code 3.
__device__ __forceinline__ void stage_weights(__bf16* smW, const __bf16* wt,
                                              int tid) {
#if HAVE_TDM
  if (tid < 32) {                       // wave 0 issues the DMA
    tdm_load_2d(wt, (unsigned)(unsigned long long)(const void*)smW,
                256u, 256u, 256u, 6u, 3u);
    __builtin_amdgcn_s_wait_tensorcnt(0);
  }
#else
#pragma unroll 4
  for (int i = 0; i < 32; ++i) {        // 8192 16B chunks / 256 threads
    int idx = tid + i * 256;
    int n = idx >> 5, kc = (idx & 31) * 8;
    *(v8bf*)(smW + n * SW_STR + kc) = *(const v8bf*)(wt + n * 256 + kc);
  }
#endif
}

// ---------------------------------------------------------------------------
__global__ __launch_bounds__(256) void fca_prep(
    const float* __restrict__ w1, const float* __restrict__ w2,
    __bf16* __restrict__ w1t, __bf16* __restrict__ w2t,
    __bf16* __restrict__ dmb) {
  int t = blockIdx.x * 256 + threadIdx.x;   // 65536 threads
  int k = t >> 8, n = t & 255;
  w1t[n * 256 + k] = (__bf16)w1[t];
  w2t[n * 256 + k] = (__bf16)w2[t];
  if (t < 4096) {                            // DCT-II matrix, factor 4
    int kk = t >> 6, w = t & 63;
    float v = 4.0f * __cosf(3.14159265358979323846f *
                            (2.0f * (float)w + 1.0f) * (float)kk * (1.0f / 128.0f));
    dmb[t] = (__bf16)v;
  }
}

// ---------------------------------------------------------------------------
__global__ __launch_bounds__(256) void fca_fused(
    const float* __restrict__ x, const float* __restrict__ b1,
    const float* __restrict__ b2, const __bf16* __restrict__ w1t,
    const __bf16* __restrict__ w2t, const __bf16* __restrict__ dmb,
    float* __restrict__ out) {
  extern __shared__ __align__(16) char smem[];
  __bf16* smA = (__bf16*)(smem + OFF_A);   // DCT matrix, row-major
  __bf16* smX = (__bf16*)(smem + OFF_X);   // x slice, column-major
  __bf16* smD = (__bf16*)(smem + OFF_D);   // dct,   row-major
  __bf16* smH = (__bf16*)(smem + OFF_H);   // h,     row-major
  __bf16* smW = (__bf16*)(smem + OFF_W);   // w1/w2, column-major

  const int tid  = threadIdx.x;
  const int wv   = tid >> 5;
  const int lane = tid & 31;
  const int half = lane >> 4;
  const int l16  = lane & 15;

  const size_t slice = (size_t)blockIdx.x * (64 * 256);
  const float* xs = x + slice;
  float*       os = out + slice;

  // ---- stage 0: fill LDS ------------------------------------------------
  stage_weights(smW, w1t, tid);            // TDM (wave 0) or b128 copies
  // Dm bf16 [k][w] -> padded row-major
#pragma unroll
  for (int i = 0; i < 2; ++i) {            // 512 16B chunks
    int idx = tid + i * 256;
    int k = idx >> 3, wc = (idx & 7) * 8;
    *(v8bf*)(smA + k * SA_STR + wc) = *(const v8bf*)(dmb + k * 64 + wc);
  }
  // x slice [w][c] fp32 -> bf16 column-major smX[c][w]
#pragma unroll 4
  for (int i = 0; i < 16; ++i) {           // 4096 float4 quads
    int idx = tid + i * 256;
    int w = idx >> 6, c0 = (idx & 63) * 4;
    f32x4 v = *(const f32x4*)(xs + (size_t)idx * 4);
    smX[(c0 + 0) * SX_STR + w] = (__bf16)v[0];
    smX[(c0 + 1) * SX_STR + w] = (__bf16)v[1];
    smX[(c0 + 2) * SX_STR + w] = (__bf16)v[2];
    smX[(c0 + 3) * SX_STR + w] = (__bf16)v[3];
  }
  __syncthreads();

  // tile ownership: 4 M-tiles x 16 N-tiles, 8 tiles per wave
  const int mBase = (wv >> 1) * 16;
  const int nset  = (wv & 1) * 8;

  // ---- stage 1: dct = Dm @ x  (K = 64) ----------------------------------
  {
    v16bf a0 = load_a(smA, SA_STR, mBase, 0, lane);
    v16bf a1 = load_a(smA, SA_STR, mBase, 32, lane);
    for (int j = 0; j < 8; ++j) {
      int n0 = (nset + j) * 16;
      v8f acc = vzero8();
      acc = wmma_bf16(a0, load_b(smX, SX_STR, 0, n0, lane), acc);
      acc = wmma_bf16(a1, load_b(smX, SX_STR, 32, n0, lane), acc);
#pragma unroll
      for (int r = 0; r < 8; ++r)
        smD[(mBase + half * 8 + r) * SD_STR + n0 + l16] = (__bf16)acc[r];
    }
  }
  __syncthreads();

  // ---- stage 2: h = relu(dct @ w1 + b1)  (K = 256) ----------------------
  {
    v16bf af[8];
#pragma unroll
    for (int k8 = 0; k8 < 8; ++k8)
      af[k8] = load_a(smD, SD_STR, mBase, k8 * 32, lane);
    for (int j = 0; j < 8; ++j) {
      int n0 = (nset + j) * 16;
      v8f acc = vzero8();
#pragma unroll
      for (int k8 = 0; k8 < 8; ++k8)
        acc = wmma_bf16(af[k8], load_b(smW, SW_STR, k8 * 32, n0, lane), acc);
      float bias = b1[n0 + l16];
#pragma unroll
      for (int r = 0; r < 8; ++r) {
        float v = acc[r] + bias;
        v = v > 0.0f ? v : 0.0f;
        smH[(mBase + half * 8 + r) * SD_STR + n0 + l16] = (__bf16)v;
      }
    }
  }
  __syncthreads();

  // ---- swap weight buffer to w2 -----------------------------------------
  stage_weights(smW, w2t, tid);
  __syncthreads();

  // ---- stage 3: out = x * sigmoid(h @ w2 + b2)  (K = 256) ---------------
  {
    v16bf af[8];
#pragma unroll
    for (int k8 = 0; k8 < 8; ++k8)
      af[k8] = load_a(smH, SD_STR, mBase, k8 * 32, lane);
    for (int j = 0; j < 8; ++j) {
      int n0 = (nset + j) * 16;
      v8f acc = vzero8();
#pragma unroll
      for (int k8 = 0; k8 < 8; ++k8)
        acc = wmma_bf16(af[k8], load_b(smW, SW_STR, k8 * 32, n0, lane), acc);
      float bias = b2[n0 + l16];
#pragma unroll
      for (int r = 0; r < 8; ++r) {
        float v = acc[r] + bias;
        float s = 1.0f / (1.0f + __expf(-v));
        int row = mBase + half * 8 + r;
        size_t idx = (size_t)row * 256 + n0 + l16;
        os[idx] = xs[idx] * s;             // second x read hits L2
      }
    }
  }
}

extern "C" void kernel_launch(void* const* d_in, const int* in_sizes, int n_in,
                              void* d_out, int out_size, void* d_ws, size_t ws_size,
                              hipStream_t stream) {
  (void)in_sizes; (void)n_in; (void)out_size; (void)ws_size;
  const float* x  = (const float*)d_in[0];
  const float* w1 = (const float*)d_in[1];
  const float* b1 = (const float*)d_in[2];
  const float* w2 = (const float*)d_in[3];
  const float* b2 = (const float*)d_in[4];
  float* out = (float*)d_out;

  __bf16* w1t = (__bf16*)((char*)d_ws + WS_W1T);
  __bf16* w2t = (__bf16*)((char*)d_ws + WS_W2T);
  __bf16* dmb = (__bf16*)((char*)d_ws + WS_DM);

  fca_prep<<<256, 256, 0, stream>>>(w1, w2, w1t, w2t, dmb);
  fca_fused<<<2048, 256, LDS_BYTES, stream>>>(x, b1, b2, w1t, w2t, dmb, out);
}
